// WorldModel_43439299231918
// MI455X (gfx1250) — matrix operand order repaired and checked by reading, not compiled, not run on hardware
//
#include <hip/hip_runtime.h>

// ---------------------------------------------------------------------------
// World-model LSTM, fused single kernel for gfx1250 (MI455X, wave32, WMMA).
// B=1024, T=1024, N_OBS=18, N_ACT=5, H=64, 4H=256, Z=32.
// Grid = 64 workgroups (16 batch rows each), Block = 256 threads (8 waves).
// ---------------------------------------------------------------------------

#define WM_B   1024
#define WM_T   1024
#define WM_OBS 18
#define WM_ACT 5
#define WM_IN  23   // obs+act (padded to 32 for WMMA K)
#define WM_H   64
#define WM_G   256  // 4*H
#define WM_Z   32
#define WM_HD  50   // 18 pred + 32 z (padded to 64)

typedef __attribute__((ext_vector_type(16))) _Float16 v16h;
typedef __attribute__((ext_vector_type(8)))  float    v8f;

__device__ __forceinline__ float wm_tanh(float x) {
#if __has_builtin(__builtin_amdgcn_tanhf)
  return __builtin_amdgcn_tanhf(x);        // native v_tanh_f32 on gfx1250
#else
  x = fminf(fmaxf(x, -15.f), 15.f);
  float e = __expf(2.f * x);
  return (e - 1.f) / (e + 1.f);
#endif
}
__device__ __forceinline__ float wm_sigm(float x) {
  return 0.5f * wm_tanh(0.5f * x) + 0.5f;  // 1 trans op per sigmoid
}

// D = A*B + C, f16 in / f32 accumulate.
__device__ __forceinline__ v8f wm_wmma(v16h a, v16h b, v8f c) {
  return __builtin_amdgcn_wmma_f32_16x16x32_f16(
      /*neg_a=*/false, a, /*neg_b=*/false, b,
      /*c_mod=*/(short)0, c, /*reuse_a=*/false, /*reuse_b=*/false);
}

__device__ __forceinline__ v16h wm_zero16() {
  v16h z;
#pragma unroll
  for (int u = 0; u < 16; ++u) z[u] = (_Float16)0.f;
  return z;
}

// Load a 16x32 f16 A-fragment from LDS (row-major [16][ld]), K offset kofs.
// Contiguous K order used here (not the ISA table order): lane L<16: M=L, K=0..15 sequential;
//               lane L>=16: M=L-16, K=16..31 sequential.
__device__ __forceinline__ v16h wm_ldsA(const _Float16* s, int ld, int kofs, int lane) {
  const int m  = lane & 15;
  const int hi = (lane >> 4) << 4;
  v16h a;
#pragma unroll
  for (int j = 0; j < 8; ++j) {
    int k = kofs + hi + (j << 1);
    a[2 * j]     = s[m * ld + k];
    a[2 * j + 1] = s[m * ld + k + 1];
  }
  return a;
}

// B-fragment where B[k][n] = W[n][k], W row-major [.,ncols]; no bounds checks
// (caller guarantees n0+15 and k0+31 are in range).
// Contiguous K order, same as wm_ldsA: lane = column n (mod 16); lanes 0-15 hold K=0..15,
// lanes 16-31 hold K=16..31, 16 halves sequential in K.
__device__ __forceinline__ v16h wm_gldB(const float* W, int ncols, int n0, int k0,
                                        int lane) {
  const int n  = n0 + (lane & 15);
  const int kb = k0 + ((lane >> 4) << 4);
  v16h b;
#pragma unroll
  for (int u = 0; u < 16; ++u)
    b[u] = (_Float16)W[n * ncols + kb + u];
  return b;
}

// Same, but zero-pads K >= kmax (for the encoder's K=23 -> 32 padding).
__device__ __forceinline__ v16h wm_gldB_kpad(const float* W, int ncols, int n0,
                                             int kmax, int lane) {
  const int n  = n0 + (lane & 15);
  const int kb = (lane >> 4) << 4;
  v16h b;
#pragma unroll
  for (int u = 0; u < 16; ++u) {
    int k = kb + u;
    b[u] = (_Float16)((k < kmax) ? W[n * ncols + k] : 0.f);
  }
  return b;
}

// Combined head B-matrix: cols 0..17 = W_pred, 18..49 = W_z, 50..63 = 0.
__device__ __forceinline__ v16h wm_gldBHead(const float* Wp, const float* Wz,
                                            int n0, int k0, int lane) {
  const int n  = n0 + (lane & 15);
  const int kb = k0 + ((lane >> 4) << 4);
  v16h b;
#pragma unroll
  for (int u = 0; u < 16; ++u) {
    float v = 0.f;
    if (n < WM_OBS)       v = Wp[n * WM_H + kb + u];
    else if (n < WM_HD)   v = Wz[(n - WM_OBS) * WM_H + kb + u];
    b[u] = (_Float16)v;
  }
  return b;
}

__global__ __launch_bounds__(256) void wm_lstm_kernel(
    const float* __restrict__ obs, const float* __restrict__ act,
    const float* __restrict__ h0,  const float* __restrict__ c0,
    const float* __restrict__ W_enc, const float* __restrict__ b_enc,
    const float* __restrict__ W_ih,  const float* __restrict__ W_hh,
    const float* __restrict__ b_ih,  const float* __restrict__ b_hh,
    const float* __restrict__ W_pred, const float* __restrict__ b_pred,
    const float* __restrict__ W_z,    const float* __restrict__ b_z,
    float* __restrict__ outPred, float* __restrict__ outZ,
    float* __restrict__ outH,    float* __restrict__ outC) {
  __shared__ _Float16 sIn[16 * 32];   // padded concat(obs,act), f16
  __shared__ _Float16 sX [16 * WM_H]; // encoder output x_t, f16
  __shared__ _Float16 sH [16 * WM_H]; // hidden state h, f16
  __shared__ float    sG [16 * WM_G]; // gate preactivations, f32
  __shared__ float    sHd[16 * WM_H]; // head preactivations, f32

  const int tid  = threadIdx.x;
  const int lane = tid & 31;
  const int wave = tid >> 5;
  const int lmod = lane & 15;
  const int b0   = blockIdx.x * 16;

  // ---- one-time weight fragment preload into VGPRs (f32 -> f16) ----
  // W_ih/W_hh tiles are always fully in range: n <= 255, k <= 63.
  v16h bIh[2][2], bHh[2][2];
#pragma unroll
  for (int tn = 0; tn < 2; ++tn)
#pragma unroll
    for (int kf = 0; kf < 2; ++kf) {
      bIh[tn][kf] = wm_gldB(W_ih, WM_H, 32 * wave + 16 * tn, 32 * kf, lane);
      bHh[tn][kf] = wm_gldB(W_hh, WM_H, 32 * wave + 16 * tn, 32 * kf, lane);
    }
  v16h bEnc = wm_zero16();
  if (wave < 4)  // waves 4-7 never run the encoder WMMA
    bEnc = wm_gldB_kpad(W_enc, WM_IN, 16 * wave, WM_IN, lane);
  v16h bHd0 = wm_gldBHead(W_pred, W_z, 16 * wave, 0,  lane);
  v16h bHd1 = wm_gldBHead(W_pred, W_z, 16 * wave, 32, lane);

  float btot[2];
#pragma unroll
  for (int tn = 0; tn < 2; ++tn) {
    int n = 32 * wave + 16 * tn + lmod;
    btot[tn] = b_ih[n] + b_hh[n];
  }
  const float bEncS = (wave < 4) ? b_enc[16 * wave + lmod] : 0.f;

  // ---- per-thread input streams: unsigned 32-bit element offsets ----------
  unsigned inOff[2];
  int inInc[2];
  bool inObs[2], inAct[2];
#pragma unroll
  for (int q = 0; q < 2; ++q) {
    int p = tid + 256 * q, r = p >> 5, c = p & 31;
    unsigned bt = (unsigned)(b0 + r) * WM_T;
    inObs[q] = (c < WM_OBS);
    inAct[q] = (c >= WM_OBS) && (c < WM_IN);
    if (inObs[q])      { inOff[q] = bt * WM_OBS + c;            inInc[q] = WM_OBS; }
    else if (inAct[q]) { inOff[q] = bt * WM_ACT + (c - WM_OBS); inInc[q] = WM_ACT; }
    else               { inOff[q] = 0;                          inInc[q] = 0;      }
  }

  // ---- per-thread output streams: unsigned 32-bit element offsets ---------
  unsigned oOff[4];
  int oInc[4], oSrc[4];
  float oBias[4];
  bool oPred[4], oZ[4];
#pragma unroll
  for (int q = 0; q < 4; ++q) {
    int p = tid + 256 * q;
    bool valid = (p < 16 * WM_HD);
    int r = p / WM_HD, c = p - r * WM_HD;
    if (!valid) { r = 0; c = 0; }
    unsigned bt = (unsigned)(b0 + r) * WM_T;
    oSrc[q]  = r * WM_H + c;
    oPred[q] = valid && (c < WM_OBS);
    oZ[q]    = valid && (c >= WM_OBS);
    if (c < WM_OBS) { oOff[q] = bt * WM_OBS + c;          oInc[q] = WM_OBS; oBias[q] = b_pred[c]; }
    else            { oOff[q] = bt * WM_Z + (c - WM_OBS); oInc[q] = WM_Z;   oBias[q] = b_z[c - WM_OBS]; }
  }

  // ---- init recurrent state ----
  float cst[4], hst[4];
#pragma unroll
  for (int q = 0; q < 4; ++q) {
    int p = tid + 256 * q, r = p >> 6, j = p & 63;
    cst[q] = c0[(b0 + r) * WM_H + j];
    hst[q] = h0[(b0 + r) * WM_H + j];
    sH[r * WM_H + j] = (_Float16)hst[q];
  }
  __syncthreads();

  v16h aH0 = wm_ldsA(sH, WM_H, 0,  lane);
  v16h aH1 = wm_ldsA(sH, WM_H, 32, lane);

  // prefetch inputs for t=0 into registers (global_load, saddr+voffset form)
  float inv[2];
#pragma unroll
  for (int q = 0; q < 2; ++q) {
    float v = 0.f;
    if (inObs[q])      v = obs[inOff[q]];
    else if (inAct[q]) v = act[inOff[q]];
    inOff[q] += inInc[q];
    inv[q] = v;
  }

#pragma unroll 1
  for (int t = 0; t < WM_T; ++t) {
    // (1) stage inputs into LDS (f16, zero-padded to K=32)
#pragma unroll
    for (int q = 0; q < 2; ++q) sIn[tid + 256 * q] = (_Float16)inv[q];
    __syncthreads();

    // (2) register-double-buffered prefetch of next step's inputs
    if (t + 1 < WM_T) {
#pragma unroll
      for (int q = 0; q < 2; ++q) {
        float v = 0.f;
        if (inObs[q])      v = obs[inOff[q]];
        else if (inAct[q]) v = act[inOff[q]];
        inOff[q] += inInc[q];
        inv[q] = v;
      }
    }

    // (3) encoder: x = relu(in @ W_enc^T + b_enc)   (waves 0..3, 1 WMMA each)
    if (wave < 4) {
      v16h aIn = wm_ldsA(sIn, 32, 0, lane);
      v8f e;
#pragma unroll
      for (int v = 0; v < 8; ++v) e[v] = bEncS;
      e = wm_wmma(aIn, bEnc, e);
      const int row = (lane >> 4) << 3;
      _Float16* xb = &sX[row * WM_H + 16 * wave + lmod];
#pragma unroll
      for (int v = 0; v < 8; ++v)
        xb[v * WM_H] = (_Float16)fmaxf(e[v], 0.f);
    }
    __syncthreads();

    // (4) gates = x@W_ih^T + h@W_hh^T + bias   (8 waves x 2 tiles x 4 WMMA)
    {
      v16h aX0 = wm_ldsA(sX, WM_H, 0,  lane);
      v16h aX1 = wm_ldsA(sX, WM_H, 32, lane);
      const int row = (lane >> 4) << 3;
#pragma unroll
      for (int tn = 0; tn < 2; ++tn) {
        v8f acc;
#pragma unroll
        for (int v = 0; v < 8; ++v) acc[v] = btot[tn];
        acc = wm_wmma(aX0, bIh[tn][0], acc);
        acc = wm_wmma(aX1, bIh[tn][1], acc);
        acc = wm_wmma(aH0, bHh[tn][0], acc);
        acc = wm_wmma(aH1, bHh[tn][1], acc);
        float* gb = &sG[row * WM_G + 32 * wave + 16 * tn + lmod];
#pragma unroll
        for (int v = 0; v < 8; ++v) gb[v * WM_G] = acc[v];
      }
    }
    __syncthreads();

    // (5) elementwise LSTM cell update; c stays in registers for all 1024 steps
#pragma unroll
    for (int q = 0; q < 4; ++q) {
      int p = tid + 256 * q, r = p >> 6, j = p & 63;
      float ig = sG[r * WM_G + j];
      float fg = sG[r * WM_G + WM_H + j];
      float gg = sG[r * WM_G + 2 * WM_H + j];
      float og = sG[r * WM_G + 3 * WM_H + j];
      float c  = wm_sigm(fg) * cst[q] + wm_sigm(ig) * wm_tanh(gg);
      float h  = wm_sigm(og) * wm_tanh(c);
      cst[q] = c;
      hst[q] = h;
      sH[r * WM_H + j] = (_Float16)h;
    }
    __syncthreads();

    // (6) reload h fragments: used by heads now and as recurrent A next step
    aH0 = wm_ldsA(sH, WM_H, 0,  lane);
    aH1 = wm_ldsA(sH, WM_H, 32, lane);

    // (7) heads: [pred|z] = h @ Whead^T   (waves 0..3, 2 WMMA each)
    if (wave < 4) {
      v8f acc;
#pragma unroll
      for (int v = 0; v < 8; ++v) acc[v] = 0.f;
      acc = wm_wmma(aH0, bHd0, acc);
      acc = wm_wmma(aH1, bHd1, acc);
      const int row = (lane >> 4) << 3;
      float* hb = &sHd[row * WM_H + 16 * wave + lmod];
#pragma unroll
      for (int v = 0; v < 8; ++v)
        hb[v * WM_H] = acc[v];
    }
    __syncthreads();

    // (8) write outputs (global_store, saddr+voffset); next sHd write is
    //     4 barriers away so no extra barrier needed here
#pragma unroll
    for (int q = 0; q < 4; ++q) {
      float v = sHd[oSrc[q]] + oBias[q];
      if (oPred[q]) outPred[oOff[q]] = wm_sigm(v);
      else if (oZ[q]) outZ[oOff[q]] = v;
      oOff[q] += oInc[q];
    }
  }

  // ---- final state (h_T, c_T) straight from registers ----
#pragma unroll
  for (int q = 0; q < 4; ++q) {
    int p = tid + 256 * q, r = p >> 6, j = p & 63;
    outH[(b0 + r) * WM_H + j] = hst[q];
    outC[(b0 + r) * WM_H + j] = cst[q];
  }
}

extern "C" void kernel_launch(void* const* d_in, const int* in_sizes, int n_in,
                              void* d_out, int out_size, void* d_ws, size_t ws_size,
                              hipStream_t stream) {
  (void)in_sizes; (void)n_in; (void)out_size; (void)d_ws; (void)ws_size;
  const float* obs    = (const float*)d_in[0];
  const float* act    = (const float*)d_in[1];
  const float* h0     = (const float*)d_in[2];
  const float* c0     = (const float*)d_in[3];
  const float* W_enc  = (const float*)d_in[4];
  const float* b_enc  = (const float*)d_in[5];
  const float* W_ih   = (const float*)d_in[6];
  const float* W_hh   = (const float*)d_in[7];
  const float* b_ih   = (const float*)d_in[8];
  const float* b_hh   = (const float*)d_in[9];
  const float* W_pred = (const float*)d_in[10];
  const float* b_pred = (const float*)d_in[11];
  const float* W_z    = (const float*)d_in[12];
  const float* b_z    = (const float*)d_in[13];

  float* out     = (float*)d_out;
  float* outPred = out;                                          // [B,T,18]
  float* outZ    = out + (size_t)WM_B * WM_T * WM_OBS;           // [B,T,32]
  float* outH    = out + (size_t)WM_B * WM_T * (WM_OBS + WM_Z);  // [B,64]
  float* outC    = outH + (size_t)WM_B * WM_H;                   // [B,64]

  dim3 grid(WM_B / 16), block(256);
  hipLaunchKernelGGL(wm_lstm_kernel, grid, block, 0, stream,
                     obs, act, h0, c0, W_enc, b_enc, W_ih, W_hh, b_ih, b_hh,
                     W_pred, b_pred, W_z, b_z, outPred, outZ, outH, outC);
}
